// LSTM_27118423507725
// MI455X (gfx1250) — compile-verified, matrix operand order checked by reading, not run on hardware
//
#include <hip/hip_runtime.h>
#include <hip/hip_bf16.h>

// ---------------- problem constants (from reference) ----------------
#define NB    64      // batch
#define TSTEPS 512    // time steps
#define DIN   256     // input dim
#define MM    512     // hidden dim
#define KC    2       // number of cells
#define OUTD  128     // fc out
#define G4M   2048    // 4*M
#define RTOT  768     // DIN + MM  (fused A reduction dim)
#define NCOLS 4096    // KC * 4M   (fused gate columns)
#define KITERS 24     // RTOT / 32
#define ASTRIDE (RTOT + 16)   // LDS row stride (bf16), keeps 16B alignment

// d_out flat offsets (floats), in reference return order
#define O_OUT  0
#define O_Z    8192            // [N,T,K]
#define O_QZ   73728
#define O_HT   139264          // [N,T,M]
#define O_HFIN 16916480        // [N,M]
#define O_CFIN 16949248

// workspace byte offsets
#define WS_WGFRAG 0                         // 24*256*32*16 bf16 = 6 MB
#define WS_WZ     (6291456)                 // 768*2 f32
#define WS_H0     (WS_WZ + 8192)            // 64*512 f32
#define WS_H1     (WS_H0 + 131072)
#define WS_C0     (WS_H1 + 131072)
#define WS_C1     (WS_C0 + 131072)

typedef __attribute__((ext_vector_type(16))) __bf16 v16bf;
typedef __attribute__((ext_vector_type(8)))  __bf16 v8bf;
typedef __attribute__((ext_vector_type(8)))  float  v8f;

__device__ __forceinline__ __bf16 f2bf(float f) {
    union { float f; unsigned u; } v; v.f = f;
    unsigned r = v.u + 0x7FFFu + ((v.u >> 16) & 1u);   // round-to-nearest-even
    union { unsigned short s; __bf16 b; } o; o.s = (unsigned short)(r >> 16);
    return o.b;
}
__device__ __forceinline__ float bf2f(__bf16 b) {
    union { unsigned short s; __bf16 b; } i; i.b = b;
    union { unsigned u; float f; } o; o.u = ((unsigned)i.s) << 16;
    return o.f;
}
__device__ __forceinline__ float sigmoidf(float x) { return 1.0f / (1.0f + __expf(-x)); }

// ------------------------------------------------------------------
// Pack [W_ih; W_hh] (fp32) -> bf16 in WMMA-B fragment-swizzled layout:
// flat = ((kIter*256 + colGroup)*32 + lane)*16 + e
// lane: col = cg*16 + (lane&15), sub = lane>>4
// e:    Klocal = sub*8 + (e<8 ? e : e+8); r = kIter*32 + Klocal
// ------------------------------------------------------------------
__global__ __launch_bounds__(256) void pack_weights(const float* __restrict__ W_ih,
                                                    const float* __restrict__ W_hh,
                                                    __bf16* __restrict__ WgFrag) {
    const unsigned total = 24u * 256u * 32u * 16u;   // 3,145,728
    for (unsigned i = blockIdx.x * blockDim.x + threadIdx.x; i < total;
         i += gridDim.x * blockDim.x) {
        int e    = i & 15;
        int lane = (i >> 4) & 31;
        int cg   = (i >> 9) & 255;
        int kI   = i >> 17;
        int sub  = lane >> 4;
        int Klocal = sub * 8 + (e < 8 ? e : e + 8);
        int r  = kI * 32 + Klocal;                 // fused reduction row
        int cp = cg * 16 + (lane & 15);            // fused gate column (0..4095)
        int k  = cp >> 11;                         // cell
        int o  = cp & 2047;                        // gate-column within cell
        float v = (r < DIN)
                      ? W_ih[(size_t)k * DIN * G4M + (size_t)r * G4M + o]
                      : W_hh[(size_t)k * MM * G4M + (size_t)(r - DIN) * G4M + o];
        WgFrag[i] = f2bf(v);
    }
}

// Fused [W_zx; W_zh] pack + zero-init of h0/c0 state
__global__ __launch_bounds__(256) void init_state(const float* __restrict__ W_zx,
                                                  const float* __restrict__ W_zh,
                                                  float* __restrict__ Wz,
                                                  float* __restrict__ h0,
                                                  float* __restrict__ c0) {
    for (int i = blockIdx.x * blockDim.x + threadIdx.x; i < NB * MM;
         i += gridDim.x * blockDim.x) {
        h0[i] = 0.0f;
        c0[i] = 0.0f;
        if (i < RTOT * KC) {
            int r = i >> 1, k = i & 1;
            Wz[i] = (r < DIN) ? W_zx[r * KC + k] : W_zh[(r - DIN) * KC + k];
        }
    }
}

// ------------------------------------------------------------------
// One recurrent step. grid = (32 m-tiles, 4 row-tiles), block = 128.
// ------------------------------------------------------------------
__global__ __launch_bounds__(128) void lstm_step(const float* __restrict__ x,
                                                 const float* __restrict__ tau_p,
                                                 const int*   __restrict__ length,
                                                 const float* __restrict__ bias,   // [K,4M]
                                                 const float* __restrict__ b_z,    // [K]
                                                 const float* __restrict__ Wz,     // [768][2]
                                                 const __bf16* __restrict__ WgFrag,
                                                 const float* __restrict__ h_in,
                                                 const float* __restrict__ c_in,
                                                 float* __restrict__ h_out,
                                                 float* __restrict__ c_out,
                                                 float* __restrict__ dout,
                                                 int t) {
    __shared__ __align__(16) __bf16 Ash[16][ASTRIDE];   // [x_t | h] rows, bf16
    __shared__ float Cbuf[8][16][16];                   // 8 gate tiles (k,gate)
    __shared__ float lpart[4][16][2];
    __shared__ float zsh[16][2], qzsh[16][2];

    const int tid = threadIdx.x;
    const int mt  = blockIdx.x;         // m-tile (0..31)
    const int n0  = blockIdx.y * 16;    // row-tile base
    const int m0  = mt * 16;

    // ---- stage A = [x_t | h] (16 x 768) into LDS as bf16 ----
    for (int idx = tid; idx < 16 * RTOT; idx += 128) {
        int row = idx / RTOT, r = idx % RTOT;
        int n = n0 + row;
        float v = (r < DIN) ? x[((size_t)n * TSTEPS + t) * DIN + r]
                            : h_in[n * MM + (r - DIN)];
        Ash[row][r] = f2bf(v);
    }
    __syncthreads();

    // ---- gating logits from the staged A tile ----
    {
        int row = tid & 15, kk = (tid >> 4) & 1, q = tid >> 5;
        float acc = 0.0f;
        int rbeg = q * (RTOT / 4), rend = rbeg + (RTOT / 4);
        for (int r = rbeg; r < rend; ++r)
            acc += bf2f(Ash[row][r]) * Wz[r * KC + kk];
        lpart[q][row][kk] = acc;
    }
    __syncthreads();
    if (tid < 16) {
        float l0 = b_z[0], l1 = b_z[1];
        for (int q = 0; q < 4; ++q) { l0 += lpart[q][tid][0]; l1 += lpart[q][tid][1]; }
        float mx = fmaxf(l0, l1);
        float e0 = __expf(l0 - mx), e1 = __expf(l1 - mx), inv = 1.0f / (e0 + e1);
        qzsh[tid][0] = e0 * inv; qzsh[tid][1] = e1 * inv;
        float it = 1.0f / tau_p[0];
        float s0 = l0 * it, s1 = l1 * it;
        mx = fmaxf(s0, s1);
        e0 = __expf(s0 - mx); e1 = __expf(s1 - mx); inv = 1.0f / (e0 + e1);
        zsh[tid][0] = e0 * inv; zsh[tid][1] = e1 * inv;
    }
    __syncthreads();

    // ---- gate GEMM: 4 waves x 2 (k,gate) tiles, K-loop of bf16 WMMA.
    //      Two K-slices per iteration with 4 independent accumulator
    //      chains -> dependent-WMMA distance of 4 hides the 5-slot
    //      bf16 WMMA RAW hazard without v_nop padding. ----
    const int wave = tid >> 5, lane = tid & 31;
    const int g0 = wave * 2, g1 = wave * 2 + 1;
    const int cg0 = (g0 >> 2) * 128 + (g0 & 3) * 32 + mt;
    const int cg1 = (g1 >> 2) * 128 + (g1 & 3) * 32 + mt;
    const int arow = lane & 15, sub = lane >> 4;

    v8f acc0a = {}, acc0b = {}, acc1a = {}, acc1b = {};
    const v16bf* bp = (const v16bf*)WgFrag;
    for (int kI = 0; kI < KITERS; kI += 2) {
        const v8bf* a0lo = (const v8bf*)&Ash[arow][kI * 32 + sub * 8];
        const v8bf* a0hi = (const v8bf*)&Ash[arow][kI * 32 + sub * 8 + 16];
        const v8bf* a1lo = (const v8bf*)&Ash[arow][(kI + 1) * 32 + sub * 8];
        const v8bf* a1hi = (const v8bf*)&Ash[arow][(kI + 1) * 32 + sub * 8 + 16];
        v16bf a0 = __builtin_shufflevector(*a0lo, *a0hi,
                                           0, 1, 2, 3, 4, 5, 6, 7,
                                           8, 9, 10, 11, 12, 13, 14, 15);
        v16bf a1 = __builtin_shufflevector(*a1lo, *a1hi,
                                           0, 1, 2, 3, 4, 5, 6, 7,
                                           8, 9, 10, 11, 12, 13, 14, 15);
        v16bf b00 = bp[(kI * 256 + cg0) * 32 + lane];
        v16bf b10 = bp[(kI * 256 + cg1) * 32 + lane];
        v16bf b01 = bp[((kI + 1) * 256 + cg0) * 32 + lane];
        v16bf b11 = bp[((kI + 1) * 256 + cg1) * 32 + lane];
        if (kI + 2 < KITERS) {   // WGP-scope prefetch of the next K-slices
            __builtin_prefetch(&bp[((kI + 2) * 256 + cg0) * 32 + lane], 0, 3);
            __builtin_prefetch(&bp[((kI + 2) * 256 + cg1) * 32 + lane], 0, 3);
        }
        acc0a = __builtin_amdgcn_wmma_f32_16x16x32_bf16(false, a0, false, b00,
                                                        (short)0, acc0a, false, false);
        acc1a = __builtin_amdgcn_wmma_f32_16x16x32_bf16(false, a0, false, b10,
                                                        (short)0, acc1a, false, false);
        acc0b = __builtin_amdgcn_wmma_f32_16x16x32_bf16(false, a1, false, b01,
                                                        (short)0, acc0b, false, false);
        acc1b = __builtin_amdgcn_wmma_f32_16x16x32_bf16(false, a1, false, b11,
                                                        (short)0, acc1b, false, false);
    }
    v8f acc0 = acc0a + acc0b;
    v8f acc1 = acc1a + acc1b;
#pragma unroll
    for (int j = 0; j < 8; ++j) {       // C layout: M = j + 8*sub, N = lane&15
        Cbuf[g0][j + 8 * sub][lane & 15] = acc0[j];
        Cbuf[g1][j + 8 * sub][lane & 15] = acc1[j];
    }
    __syncthreads();

    // ---- activations, mixture, masking, state update ----
    for (int idx = tid; idx < 256; idx += 128) {
        int rl = idx >> 4, ml = idx & 15;
        int n = n0 + rl, m = m0 + ml;
        float cold = c_in[n * MM + m];
        float hold = h_in[n * MM + m];
        float hnew = 0.0f, cnew = 0.0f;
#pragma unroll
        for (int k = 0; k < KC; ++k) {
            float ig = Cbuf[k * 4 + 0][rl][ml] + bias[k * G4M + 0 * MM + m];
            float fg = Cbuf[k * 4 + 1][rl][ml] + bias[k * G4M + 1 * MM + m];
            float gg = Cbuf[k * 4 + 2][rl][ml] + bias[k * G4M + 2 * MM + m];
            float og = Cbuf[k * 4 + 3][rl][ml] + bias[k * G4M + 3 * MM + m];
            float ck = sigmoidf(fg) * cold + sigmoidf(ig) * tanhf(gg);
            float hk = sigmoidf(og) * tanhf(ck);
            float zk = zsh[rl][k];
            hnew += zk * hk;
            cnew += zk * ck;
        }
        float mask = (t < length[n]) ? 1.0f : 0.0f;
        hnew = hnew * mask + hold * (1.0f - mask);
        cnew = cnew * mask + cold * (1.0f - mask);
        h_out[n * MM + m] = hnew;
        c_out[n * MM + m] = cnew;
        dout[O_HT + ((size_t)n * TSTEPS + t) * MM + m] = hnew;
    }
    if (mt == 0 && tid < 32) {
        int rl = tid >> 1, k = tid & 1;
        int n = n0 + rl;
        dout[O_Z  + ((size_t)n * TSTEPS + t) * KC + k] = zsh[rl][k];
        dout[O_QZ + ((size_t)n * TSTEPS + t) * KC + k] = qzsh[rl][k];
    }
}

// ------------------------------------------------------------------
// output = h_T @ fc_W + fc_b  (bf16 WMMA), plus h_t/c_t copies.
// 1 block x 128 threads: wave w -> row-tile w, loops 8 col-tiles.
// ------------------------------------------------------------------
__global__ __launch_bounds__(128) void final_fc(const float* __restrict__ hfin,
                                                const float* __restrict__ cfin,
                                                const float* __restrict__ fcW,
                                                const float* __restrict__ fcb,
                                                float* __restrict__ dout) {
    const int tid = threadIdx.x, wave = tid >> 5, lane = tid & 31;
    const int rl = lane & 15, sub = lane >> 4;
    const int n = wave * 16 + rl;

    for (int ct = 0; ct < OUTD / 16; ++ct) {
        v8f acc = {};
        for (int kI = 0; kI < MM / 32; ++kI) {
            v16bf a, b;
#pragma unroll
            for (int e = 0; e < 16; ++e) {
                int K = kI * 32 + sub * 8 + (e < 8 ? e : e + 8);
                a[e] = f2bf(hfin[n * MM + K]);
                b[e] = f2bf(fcW[K * OUTD + ct * 16 + (lane & 15)]);
            }
            acc = __builtin_amdgcn_wmma_f32_16x16x32_bf16(false, a, false, b,
                                                          (short)0, acc, false, false);
        }
#pragma unroll
        for (int j = 0; j < 8; ++j) {
            int M = j + 8 * sub, c = ct * 16 + (lane & 15);
            int row = wave * 16 + M;
            dout[O_OUT + row * OUTD + c] = acc[j] + fcb[c];
        }
    }
    for (int i = tid; i < NB * MM; i += 128) {
        dout[O_HFIN + i] = hfin[i];
        dout[O_CFIN + i] = cfin[i];
    }
}

// ------------------------------------------------------------------
extern "C" void kernel_launch(void* const* d_in, const int* in_sizes, int n_in,
                              void* d_out, int out_size, void* d_ws, size_t ws_size,
                              hipStream_t stream) {
    (void)in_sizes; (void)n_in; (void)out_size; (void)ws_size;
    const float* x      = (const float*)d_in[0];
    const float* tau    = (const float*)d_in[1];
    const int*   length = (const int*)  d_in[2];
    const float* W_ih   = (const float*)d_in[3];
    const float* W_hh   = (const float*)d_in[4];
    const float* bias   = (const float*)d_in[5];
    const float* W_zx   = (const float*)d_in[6];
    const float* W_zh   = (const float*)d_in[7];
    const float* b_z    = (const float*)d_in[8];
    const float* fc_W   = (const float*)d_in[9];
    const float* fc_b   = (const float*)d_in[10];
    float* dout = (float*)d_out;

    char* ws = (char*)d_ws;
    __bf16* WgFrag = (__bf16*)(ws + WS_WGFRAG);
    float*  Wz     = (float*)(ws + WS_WZ);
    float*  hbuf[2] = { (float*)(ws + WS_H0), (float*)(ws + WS_H1) };
    float*  cbuf[2] = { (float*)(ws + WS_C0), (float*)(ws + WS_C1) };

    pack_weights<<<4096, 256, 0, stream>>>(W_ih, W_hh, WgFrag);
    init_state<<<128, 256, 0, stream>>>(W_zx, W_zh, Wz, hbuf[0], cbuf[0]);

    for (int t = 0; t < TSTEPS; ++t) {
        const float* hin = hbuf[t & 1];
        const float* cin = cbuf[t & 1];
        float* hout = hbuf[(t + 1) & 1];
        float* cout = cbuf[(t + 1) & 1];
        lstm_step<<<dim3(32, 4), 128, 0, stream>>>(x, tau, length, bias, b_z, Wz,
                                                   WgFrag, hin, cin, hout, cout,
                                                   dout, t);
    }
    // T is even -> final state lands back in buffer 0
    final_fc<<<1, 128, 0, stream>>>(hbuf[0], cbuf[0], fc_W, fc_b, dout);
}
